// Attention_45432164057400
// MI455X (gfx1250) — matrix operand order repaired and checked by reading, not compiled, not run on hardware
//
#include <hip/hip_runtime.h>
#include <hip/hip_bf16.h>
#include <math.h>

// ---------------------------------------------------------------------------
// Attention kernel for MI455X (gfx1250, wave32, WMMA + async global->LDS).
//   att1 = enc @ W_enc + b_enc              [B,P,A]   (bf16 WMMA, f32 acc)
//   att2 = h @ W_dec + b_dec                [B,A]     (tiny, VALU)
//   att  = relu(att1 + att2) . W_full + b_full  -> softmax over P -> alpha
//   awe  = sum_p enc[b,p,:] * alpha[b,p]    [B,E]
// ---------------------------------------------------------------------------

typedef __attribute__((ext_vector_type(16))) __bf16 v16bf;
typedef __attribute__((ext_vector_type(8)))  __bf16 v8bf;
typedef __attribute__((ext_vector_type(8)))  float  v8f;
typedef __attribute__((ext_vector_type(4)))  float  v4f;

#define B_ 256
#define P_ 196
#define E_ 2048
#define D_ 512
#define A_ 512
#define PSTRIDE 256   // padded att row stride in workspace
#define NT 64         // K tiles in main GEMM (E_/32)
#define PIPE 3        // async pipeline depth
#define AROW 36       // LDS A-tile row stride in floats (144B = 9*16B, bank-spread)

// Async DMA: copy 16 bytes global -> LDS per lane, tracked by ASYNCcnt.
__device__ __forceinline__ void async_copy_b128(unsigned lds_addr, const void* gaddr) {
  asm volatile("global_load_async_to_lds_b128 %0, %1, off"
               :: "v"(lds_addr), "v"(gaddr) : "memory");
}

// ---------------------------------------------------------------------------
// Kernel 1: W_enc [E][A] fp32 -> W_encT [A][E] bf16 (coalesced writes)
// ---------------------------------------------------------------------------
__global__ __launch_bounds__(256)
void k_wenc_t(const float* __restrict__ W_enc, __bf16* __restrict__ Wt) {
  int idx = blockIdx.x * 256 + threadIdx.x;   // 0 .. E_*A_-1, idx = a*E_+e
  int e = idx % E_;
  int a = idx / E_;
  Wt[idx] = (__bf16)W_enc[e * A_ + a];
}

// ---------------------------------------------------------------------------
// Kernel 2: att2[b][a] = sum_d h[b][d] * W_dec[d][a] + b_dec[a]  (134 MFLOP)
// ---------------------------------------------------------------------------
__global__ __launch_bounds__(256)
void k_att2(const float* __restrict__ h, const float* __restrict__ W_dec,
            const float* __restrict__ b_dec, float* __restrict__ att2) {
  int idx = blockIdx.x * 256 + threadIdx.x;   // b*A_ + a
  int b = idx / A_, a = idx % A_;
  const float* hr = h + b * D_;
  float s = b_dec[a];
#pragma unroll 4
  for (int d = 0; d < D_; ++d)
    s = fmaf(hr[d], W_dec[d * A_ + a], s);
  att2[idx] = s;
}

// ---------------------------------------------------------------------------
// Kernel 3: main fused WMMA kernel.
// Block = 256 threads (8 waves). Tile: 32 P-rows x 512 A-cols, K = E_ = 2048.
// A-tile (32x32 fp32) streamed global->LDS via async DMA, 3-deep pipeline.
// Each wave owns 4 N-tiles (16 cols); 2 M-tiles of 16 rows -> 8 WMMA / K-step.
// Per K-step: issue all 8 B global_load_b128 first, build A frags (LDS+cvt)
// under that latency, then run the 8 WMMAs with staged loadcnt waits.
// ---------------------------------------------------------------------------
__global__ __launch_bounds__(256)
void k_att_main(const float* __restrict__ enc, const __bf16* __restrict__ Wt,
                const float* __restrict__ b_enc, const float* __restrict__ att2,
                const float* __restrict__ W_full, const float* __restrict__ b_full,
                float* __restrict__ att) {
  const int b    = blockIdx.y;
  const int p0   = blockIdx.x * 32;
  const int tid  = threadIdx.x;
  const int wave = tid >> 5;
  const int lane = tid & 31;
  const int h    = lane >> 4;   // lane-half: selects K sub-chunk (ISA layout)
  const int ln   = lane & 15;

  // A-tiles: PIPE buffers of 32 rows x 32 K fp32, row stride AROW floats.
  __shared__ __attribute__((aligned(16))) float lds_a[PIPE][32 * AROW];
  __shared__ float lds_red[8][32];

  // Per-lane column metadata: this lane's N per tile is ncol[j] = tile_base + ln.
  float bias[4], wf[4];
  const __bf16* bptr[4];
  const float* att2b = att2 + b * A_;
#pragma unroll
  for (int j = 0; j < 4; ++j) {
    int n   = (wave * 4 + j) * 16 + ln;
    bias[j] = b_enc[n] + att2b[n];
    wf[j]   = W_full[n];
    bptr[j] = Wt + (size_t)n * E_ + 16 * h;   // + k0 per iteration
  }

  // Async staging mapping: thread t copies one b128 (4 floats):
  //   row = t>>3 (0..31), col4 = t&7  -> global 128B-coalesced per row.
  const float* encb = enc + (size_t)b * P_ * E_;
  const int srow = tid >> 3, scol4 = tid & 7;
  int pr = p0 + srow; if (pr > P_ - 1) pr = P_ - 1;   // clamp (dup rows discarded)
  const float* gbase = encb + (size_t)pr * E_ + scol4 * 4;
  unsigned ldsdst[PIPE];
#pragma unroll
  for (int pp = 0; pp < PIPE; ++pp)
    ldsdst[pp] = (unsigned)(size_t)(&lds_a[pp][srow * AROW + scol4 * 4]);

  // Prologue: tiles 0 and 1 in flight.
  async_copy_b128(ldsdst[0], gbase);
  async_copy_b128(ldsdst[1], gbase + 32);

  v8f acc[2][4] = {};

  for (int it = 0; it < NT; ++it) {
    const int cur = it % PIPE;
    const int nxt = (it + 2) % PIPE;
    const int k0  = it * 32;

    // Release buf[nxt] (last read at iter it-1; WMMA forced s_wait_dscnt first).
    __syncthreads();
    if (it + 2 < NT) {
      async_copy_b128(ldsdst[nxt], gbase + (size_t)(it + 2) * 32);
      asm volatile("s_wait_asynccnt 0x2" ::: "memory");   // my part of tile it landed
    } else if (it + 1 < NT) {
      asm volatile("s_wait_asynccnt 0x1" ::: "memory");
    } else {
      asm volatile("s_wait_asynccnt 0x0" ::: "memory");
    }
    __syncthreads();   // all waves' parts of tile it landed

    // 1) Issue ALL B-fragment loads first (8 global_load_b128 in flight).
    //    B layout used here: lane N = ln, K = 16h + i (contiguous thanks to
    //    the [A][E] bf16 transpose) -> two b128 loads per N-tile.
    v8bf blo[4], bhi[4];
#pragma unroll
    for (int j = 0; j < 4; ++j) {
      const __bf16* bp = bptr[j] + k0;
      blo[j] = *(const v8bf*)(bp);
      bhi[j] = *(const v8bf*)(bp + 8);
    }

    // 2) Build A fragments under the global-load latency.
    //    A layout: lane M = ln, elements 0..15 -> K in [16h,16h+16)
    //    (matches B).  fp32->bf16 via v_cvt_pk_bf16_f32.
    v16bf afrag[2];
#pragma unroll
    for (int mt = 0; mt < 2; ++mt) {
      const float* ar = &lds_a[cur][(mt * 16 + ln) * AROW + 16 * h];
      v4f c0 = *(const v4f*)(ar);
      v4f c1 = *(const v4f*)(ar + 4);
      v4f c2 = *(const v4f*)(ar + 8);
      v4f c3 = *(const v4f*)(ar + 12);
      v16bf f;
#pragma unroll
      for (int i = 0; i < 4; ++i) {
        f[i]      = (__bf16)c0[i];
        f[i + 4]  = (__bf16)c1[i];
        f[i + 8]  = (__bf16)c2[i];
        f[i + 12] = (__bf16)c3[i];
      }
      afrag[mt] = f;
    }

    // 3) Matrix ops (compiler staggers s_wait_loadcnt per B pair).
#pragma unroll
    for (int j = 0; j < 4; ++j) {
      v16bf bfrag = __builtin_shufflevector(blo[j], bhi[j],
          0, 1, 2, 3, 4, 5, 6, 7, 8, 9, 10, 11, 12, 13, 14, 15);
#pragma unroll
      for (int mt = 0; mt < 2; ++mt)
        acc[mt][j] = __builtin_amdgcn_wmma_f32_16x16x32_bf16(
            false, afrag[mt], false, bfrag, (short)0, acc[mt][j], false, false);
    }
  }

  // Epilogue: relu(att1 + bias) . W_full, reduced over all 512 A-cols.
  // C layout: element v of acc <-> (M = mt*16 + v + 8h, N = ln).
  float part[2][8];
#pragma unroll
  for (int mt = 0; mt < 2; ++mt)
#pragma unroll
    for (int v = 0; v < 8; ++v) part[mt][v] = 0.f;

#pragma unroll
  for (int j = 0; j < 4; ++j)
#pragma unroll
    for (int mt = 0; mt < 2; ++mt)
#pragma unroll
      for (int v = 0; v < 8; ++v) {
        float x = acc[mt][j][v] + bias[j];
        x = x > 0.f ? x : 0.f;
        part[mt][v] = fmaf(x, wf[j], part[mt][v]);
      }

  // Butterfly reduce over the 16 N-lanes (bit4 never flips: halves independent).
#pragma unroll
  for (int mt = 0; mt < 2; ++mt)
#pragma unroll
    for (int v = 0; v < 8; ++v) {
      float s = part[mt][v];
      s += __shfl_xor(s, 1, 32);
      s += __shfl_xor(s, 2, 32);
      s += __shfl_xor(s, 4, 32);
      s += __shfl_xor(s, 8, 32);
      part[mt][v] = s;
    }
  if (ln == 0) {
#pragma unroll
    for (int mt = 0; mt < 2; ++mt)
#pragma unroll
      for (int v = 0; v < 8; ++v)
        lds_red[wave][mt * 16 + h * 8 + v] = part[mt][v];
  }
  __syncthreads();
  if (tid < 32) {
    float s = b_full[0];
#pragma unroll
    for (int w = 0; w < 8; ++w) s += lds_red[w][tid];
    int p = p0 + tid;
    if (p < P_) att[b * PSTRIDE + p] = s;
  }
}

// ---------------------------------------------------------------------------
// Kernel 4: softmax over P per batch row.
// ---------------------------------------------------------------------------
__global__ __launch_bounds__(256)
void k_softmax(const float* __restrict__ att, float* __restrict__ alpha) {
  int b = blockIdx.x, tid = threadIdx.x;
  __shared__ float red[256];
  float v = (tid < P_) ? att[b * PSTRIDE + tid] : -3.0e38f;
  red[tid] = v;
  __syncthreads();
  for (int s = 128; s > 0; s >>= 1) {
    if (tid < s) red[tid] = fmaxf(red[tid], red[tid + s]);
    __syncthreads();
  }
  float m = red[0];
  __syncthreads();
  float e = (tid < P_) ? __expf(v - m) : 0.f;
  red[tid] = e;
  __syncthreads();
  for (int s = 128; s > 0; s >>= 1) {
    if (tid < s) red[tid] += red[tid + s];
    __syncthreads();
  }
  float inv = 1.f / red[0];
  if (tid < P_) alpha[b * P_ + tid] = e * inv;
}

// ---------------------------------------------------------------------------
// Kernel 5: awe[b][e] = sum_p enc[b][p][e] * alpha[b][p]  (single 411MB sweep)
// Grid (E_/1024, B_), 256 threads, v4f per thread, non-temporal loads (TH=NT)
// so the streaming encoder read does not evict W_encT/L2 working sets.
// ---------------------------------------------------------------------------
__global__ __launch_bounds__(256)
void k_awe(const float* __restrict__ enc, const float* __restrict__ alpha,
           float* __restrict__ awe) {
  int b = blockIdx.y, tid = threadIdx.x;
  __shared__ float s_alpha[P_];
  if (tid < P_) s_alpha[tid] = alpha[b * P_ + tid];
  __syncthreads();
  const int e4 = blockIdx.x * 256 + tid;               // v4f index within row
  const v4f* encb = (const v4f*)enc + (size_t)b * P_ * (E_ / 4) + e4;
  v4f acc = {0.f, 0.f, 0.f, 0.f};
#pragma unroll 4
  for (int p = 0; p < P_; ++p) {
    v4f v = __builtin_nontemporal_load(&encb[(size_t)p * (E_ / 4)]);
    acc += s_alpha[p] * v;
  }
  ((v4f*)awe)[(size_t)b * (E_ / 4) + e4] = acc;
}

// ---------------------------------------------------------------------------
extern "C" void kernel_launch(void* const* d_in, const int* in_sizes, int n_in,
                              void* d_out, int out_size, void* d_ws, size_t ws_size,
                              hipStream_t stream) {
  const float* enc    = (const float*)d_in[0];
  const float* h      = (const float*)d_in[1];
  const float* W_enc  = (const float*)d_in[2];
  const float* b_enc  = (const float*)d_in[3];
  const float* W_dec  = (const float*)d_in[4];
  const float* b_dec  = (const float*)d_in[5];
  const float* W_full = (const float*)d_in[6];
  const float* b_full = (const float*)d_in[7];

  // Workspace layout
  char*   ws   = (char*)d_ws;
  __bf16* Wt   = (__bf16*)ws;                                     // 2 MB
  float*  att2 = (float*)(ws + (size_t)A_ * E_ * 2);              // 512 KB
  float*  att  = (float*)(ws + (size_t)A_ * E_ * 2
                             + (size_t)B_ * A_ * 4);              // 256 KB

  float* awe   = (float*)d_out;                    // [B,E]
  float* alpha = (float*)d_out + (size_t)B_ * E_;  // [B,P]

  k_wenc_t  <<<dim3((E_ * A_) / 256),      dim3(256), 0, stream>>>(W_enc, Wt);
  k_att2    <<<dim3((B_ * A_) / 256),      dim3(256), 0, stream>>>(h, W_dec, b_dec, att2);
  k_att_main<<<dim3((P_ + 31) / 32, B_),   dim3(256), 0, stream>>>(enc, Wt, b_enc, att2,
                                                                   W_full, b_full, att);
  k_softmax <<<dim3(B_),                   dim3(256), 0, stream>>>(att, alpha);
  k_awe     <<<dim3(E_ / 1024, B_),        dim3(256), 0, stream>>>(enc, alpha, awe);
}
